// TransformerDecoder_2456721293390
// MI455X (gfx1250) — matrix operand order repaired and checked by reading, not compiled, not run on hardware
//
#include <hip/hip_runtime.h>
#include <hip/hip_bf16.h>
#include <math.h>

typedef __bf16 bf16_t;
typedef __attribute__((ext_vector_type(16))) __bf16 v16bf;
typedef __attribute__((ext_vector_type(8)))  float  v8f;

static constexpr int L_ = 6, D_ = 1024, H_ = 16, DFF_ = 4096, S_ = 512, M_ = 512, B_ = 16, HD_ = 64;
static constexpr float SCALE_ = 0.125f;     // 1/sqrt(64)
static constexpr float NEGINF = -1e9f;
#define LN_EPS 1e-5f

union FragB { v16bf v; uint4 q[2]; bf16_t e[16]; };

// LDS byte offset of a __shared__ pointer: flat LDS addresses carry the LDS
// offset in addr[31:0] (CDNA5 flat aperture mapping), so truncation works.
__device__ __forceinline__ unsigned lds_off(const void* p) {
  return (unsigned)(unsigned long long)p;
}

// async global->LDS b128 copy per lane; INST_OFFSET applies to both sides.
__device__ __forceinline__ void async_b128(unsigned ldsByteOff, const void* gaddr) {
  asm volatile("global_load_async_to_lds_b128 %0, %1, off" :: "v"(ldsByteOff), "v"(gaddr) : "memory");
}
__device__ __forceinline__ void async_b128_o16(unsigned ldsByteOff, const void* gaddr) {
  asm volatile("global_load_async_to_lds_b128 %0, %1, off offset:16" :: "v"(ldsByteOff), "v"(gaddr) : "memory");
}
__device__ __forceinline__ void wait_async0() {
  asm volatile("s_wait_asynccnt 0" ::: "memory");
}
__device__ __forceinline__ void wait_ds0() {
  asm volatile("s_wait_dscnt 0" ::: "memory");
}

// ---------------- elementwise / layout kernels ----------------

__global__ void k_f32_to_bf16(const float* __restrict__ in, bf16_t* __restrict__ out, int n) {
  int i = blockIdx.x * 256 + threadIdx.x;
  if (i < n) out[i] = (bf16_t)in[i];
}

// tgt (S,B,D) f32 -> x (B,S,D) f32
__global__ void k_tgt_to_x(const float* __restrict__ tgt, float* __restrict__ x) {
  int i = blockIdx.x * 256 + threadIdx.x;        // over S*B*D
  int d = i % D_; int sb = i / D_;
  int b = sb % B_; int s = sb / B_;
  x[((size_t)b * S_ + s) * D_ + d] = tgt[i];
}

// memory (M,B,D) f32 -> (B,M,D) bf16
__global__ void k_mem_to_bsd(const float* __restrict__ mem, bf16_t* __restrict__ out) {
  int i = blockIdx.x * 256 + threadIdx.x;        // over M*B*D
  int d = i % D_; int mb = i / D_;
  int b = mb % B_; int m = mb / B_;
  out[((size_t)b * M_ + m) * D_ + d] = (bf16_t)mem[i];
}

// one block (256 thr) per row of 1024; fp32 in, bf16 normalized out
__global__ __launch_bounds__(256) void k_layernorm(const float* __restrict__ x,
                                                   const float* __restrict__ g,
                                                   const float* __restrict__ be,
                                                   bf16_t* __restrict__ out) {
  int row = blockIdx.x;
  const float* xr = x + (size_t)row * D_;
  float v[4]; float s = 0.f, s2 = 0.f;
  for (int k = 0; k < 4; ++k) { v[k] = xr[threadIdx.x + k * 256]; s += v[k]; s2 += v[k] * v[k]; }
  for (int off = 16; off >= 1; off >>= 1) { s += __shfl_xor(s, off, 32); s2 += __shfl_xor(s2, off, 32); }
  __shared__ float red[2][8];
  int wv = threadIdx.x >> 5, ln = threadIdx.x & 31;
  if (ln == 0) { red[0][wv] = s; red[1][wv] = s2; }
  __syncthreads();
  float ts = 0.f, ts2 = 0.f;
  for (int i = 0; i < 8; ++i) { ts += red[0][i]; ts2 += red[1][i]; }
  float mean = ts * (1.f / D_);
  float var  = ts2 * (1.f / D_) - mean * mean;
  float inv  = rsqrtf(var + LN_EPS);
  for (int k = 0; k < 4; ++k) {
    int c = threadIdx.x + k * 256;
    out[(size_t)row * D_ + c] = (bf16_t)(g[c] * (v[k] - mean) * inv + be[c]);
  }
}

// ---------------- WMMA GEMM:  C(m,n) = sum_k A(m,k)*W(n,k) + bias[n] ----------------
// A: Mrows x K bf16 row-major. W: N x K bf16 row-major (i.e. h @ W.T).
// Block tile 128x128, 8 waves, wave tile 32x64 (2x4 WMMA 16x16x32 accumulators).
// LDS double-buffered; tiles staged with GLOBAL_LOAD_ASYNC_TO_LDS_B128 (ASYNCcnt),
// one workgroup barrier per K-step.

template <bool RELU, bool RESID>
__global__ __launch_bounds__(256) void k_gemm(const bf16_t* __restrict__ A,
                                              const bf16_t* __restrict__ W,
                                              const float* __restrict__ bias,
                                              const float* __restrict__ resid,
                                              void* __restrict__ outp, int K, int N) {
  constexpr int LDA = 40;                       // bf16 row stride in LDS (pad vs banks)
  constexpr int BUF = 128 * LDA;                // halfs per matrix per stage
  __shared__ __align__(16) bf16_t As[2 * BUF];
  __shared__ __align__(16) bf16_t Ws[2 * BUF];
  const int bM = blockIdx.y * 128, bN = blockIdx.x * 128;
  const int tid = threadIdx.x, lane = tid & 31, wv = tid >> 5;
  const int wm = (wv >> 1) * 32, wn = (wv & 1) * 64;
  const int g = lane >> 4, r = lane & 15;

  v8f acc[2][4];
  for (int i = 0; i < 2; ++i)
    for (int j = 0; j < 4; ++j)
      for (int k = 0; k < 8; ++k) acc[i][j][k] = 0.f;

  const int ldr = tid >> 1;                     // 0..127 (tile row this thread stages)
  const int ldc = (tid & 1) * 16;               // 0 / 16 (16-half column chunk)
  const bf16_t* gA = A + (size_t)(bM + ldr) * K + ldc;
  const bf16_t* gW = W + (size_t)(bN + ldr) * K + ldc;
  const unsigned sA = lds_off(As + ldr * LDA + ldc);
  const unsigned sW = lds_off(Ws + ldr * LDA + ldc);

  auto issue = [&](int k0, int buf) {
    unsigned la = sA + (unsigned)buf * BUF * 2;  // bytes
    unsigned lw = sW + (unsigned)buf * BUF * 2;
    async_b128(la, gA + k0);
    async_b128_o16(la, gA + k0);
    async_b128(lw, gW + k0);
    async_b128_o16(lw, gW + k0);
  };

  issue(0, 0);
  int buf = 0;
  for (int k0 = 0; k0 < K; k0 += 32) {
    wait_async0();                              // this wave's current tile resident
    __syncthreads();                            // all tiles resident; prev reads done
    if (k0 + 32 < K) issue(k0 + 32, buf ^ 1);   // prefetch next stage under compute

    const bf16_t* Ab = As + buf * BUF;
    const bf16_t* Wb = Ws + buf * BUF;
    FragB af[2], bfr[4];
    for (int mi = 0; mi < 2; ++mi) {            // A frag: lane holds row (lane&15), split K halves
      const bf16_t* p = Ab + (wm + mi * 16 + r) * LDA;
      af[mi].q[0] = *(const uint4*)(p + g * 8);
      af[mi].q[1] = *(const uint4*)(p + 16 + g * 8);
    }
    for (int ni = 0; ni < 4; ++ni) {            // B frag: lane holds col n=(lane&15), split K halves
      const bf16_t* p = Wb + (wn + ni * 16 + r) * LDA;
      bfr[ni].q[0] = *(const uint4*)(p + g * 8);
      bfr[ni].q[1] = *(const uint4*)(p + 16 + g * 8);
    }
    for (int mi = 0; mi < 2; ++mi)
      for (int ni = 0; ni < 4; ++ni)
        acc[mi][ni] = __builtin_amdgcn_wmma_f32_16x16x32_bf16(
            false, af[mi].v, false, bfr[ni].v, (short)0, acc[mi][ni], false, false);
    buf ^= 1;
  }

  for (int mi = 0; mi < 2; ++mi)
    for (int ni = 0; ni < 4; ++ni) {
      int n = bN + wn + ni * 16 + r;
      float bval = bias ? bias[n] : 0.f;
      for (int rr = 0; rr < 8; ++rr) {
        int m = bM + wm + mi * 16 + rr + g * 8;
        float val = acc[mi][ni][rr] + bval;
        if (RELU) val = fmaxf(val, 0.f);
        size_t idx = (size_t)m * N + n;
        if (RESID) ((float*)outp)[idx] = resid[idx] + val;
        else       ((bf16_t*)outp)[idx] = (bf16_t)val;
      }
    }
}

// ---------------- flash attention: 1 wave per (b,h,16-query tile) ----------------
// Q/K/V bf16 row-major token buffers with column strides qStride/kStride and head offset h*64.
// CAUSAL=true: tril mask; else key-padding byte mask (B,seqK).
// V tiles staged into wave-private LDS with async copies overlapping score WMMAs + softmax.

template <bool CAUSAL>
__global__ __launch_bounds__(256) void k_attn(const bf16_t* __restrict__ Qb,
                                              const bf16_t* __restrict__ Kb,
                                              const bf16_t* __restrict__ Vb,
                                              const unsigned char* __restrict__ mask,
                                              bf16_t* __restrict__ Ob,
                                              int qStride, int kStride, int seqK) {
  __shared__ __align__(16) bf16_t Ps[8][16 * 32]; // per-wave P relayout (C-layout -> A-frag)
  __shared__ __align__(16) bf16_t Vs[8][32 * 64]; // per-wave staged V tile
  const int tid = threadIdx.x, lane = tid & 31, wv = tid >> 5;
  const int wid = blockIdx.x * 8 + wv;          // 8192 waves total
  const int qt = wid & 31;                      // S/16 = 32 query tiles
  const int h  = (wid >> 5) & 15;
  const int b  = wid >> 9;
  const int g = lane >> 4, r = lane & 15;

  FragB aq[2];
  {
    const bf16_t* qrow = Qb + (size_t)(b * S_ + qt * 16 + r) * qStride + h * HD_;
    for (int c = 0; c < 2; ++c) {
      aq[c].q[0] = *(const uint4*)(qrow + c * 32 + g * 8);
      aq[c].q[1] = *(const uint4*)(qrow + c * 32 + 16 + g * 8);
    }
  }

  v8f o[4];
  for (int c = 0; c < 4; ++c)
    for (int k = 0; k < 8; ++k) o[c][k] = 0.f;
  float mrow[8], lrow[8];
  for (int i = 0; i < 8; ++i) { mrow[i] = -1e30f; lrow[i] = 0.f; }

  const int nIter = CAUSAL ? (qt * 16 + 16 + 31) / 32 : seqK / 32;
  bf16_t* ps = Ps[wv];
  bf16_t* vs = Vs[wv];
  const unsigned vsOff = lds_off(vs + lane * 64);

  for (int it = 0; it < nIter; ++it) {
    const int kk0 = it * 32;
    // ---- stage V tile (32 keys x 64) asynchronously; 1 row per lane ----
    {
      const bf16_t* gv = Vb + (size_t)(b * seqK + kk0 + lane) * kStride + h * HD_;
#pragma unroll
      for (int u = 0; u < 4; ++u) {
        async_b128(vsOff + u * 32, gv + u * 16);
        async_b128_o16(vsOff + u * 32, gv + u * 16);
      }
    }
    // ---- scores: S(q,key) = Q . K, two 16-key n-tiles ----
    v8f st[2];
    for (int nt = 0; nt < 2; ++nt) {
      FragB bk[2];
      const bf16_t* krow = Kb + (size_t)(b * seqK + kk0 + nt * 16 + r) * kStride + h * HD_;
      for (int c = 0; c < 2; ++c) {
        bk[c].q[0] = *(const uint4*)(krow + c * 32 + g * 8);
        bk[c].q[1] = *(const uint4*)(krow + c * 32 + 16 + g * 8);
      }
      v8f s;
      for (int k = 0; k < 8; ++k) s[k] = 0.f;
      s = __builtin_amdgcn_wmma_f32_16x16x32_bf16(false, aq[0].v, false, bk[0].v, (short)0, s, false, false);
      s = __builtin_amdgcn_wmma_f32_16x16x32_bf16(false, aq[1].v, false, bk[1].v, (short)0, s, false, false);
      st[nt] = s;
    }
    // ---- bias + online softmax (row = 16 lanes of a half-wave) ----
    float p0a[8], p1a[8];
    for (int rr = 0; rr < 8; ++rr) {
      int qrow = qt * 16 + rr + g * 8;
      float v0 = st[0][rr] * SCALE_, v1 = st[1][rr] * SCALE_;
      int key0 = kk0 + r, key1 = kk0 + 16 + r;
      if (CAUSAL) {
        if (key0 > qrow) v0 = NEGINF;
        if (key1 > qrow) v1 = NEGINF;
      } else {
        if (mask[b * seqK + key0]) v0 = NEGINF;
        if (mask[b * seqK + key1]) v1 = NEGINF;
      }
      float tm = fmaxf(v0, v1);
      for (int off = 8; off >= 1; off >>= 1) tm = fmaxf(tm, __shfl_xor(tm, off, 16));
      float mnew = fmaxf(mrow[rr], tm);
      float corr = __expf(mrow[rr] - mnew);
      float p0 = __expf(v0 - mnew), p1 = __expf(v1 - mnew);
      float psum = p0 + p1;
      for (int off = 8; off >= 1; off >>= 1) psum += __shfl_xor(psum, off, 16);
      lrow[rr] = lrow[rr] * corr + psum;
      mrow[rr] = mnew;
      for (int c = 0; c < 4; ++c) o[c][rr] *= corr;
      p0a[rr] = p0; p1a[rr] = p1;
    }
    // ---- P to LDS (relayout C-format -> A-frag format) ----
    for (int rr = 0; rr < 8; ++rr) {
      int m = rr + g * 8;
      ps[m * 32 + r]      = (bf16_t)p0a[rr];
      ps[m * 32 + 16 + r] = (bf16_t)p1a[rr];
    }
    wait_async0();                              // V tile resident (wave-private)
    wait_ds0();                                 // P relayout visible
    // ---- O += P(16x32) @ V(32x64) ----
    FragB ap;
    ap.q[0] = *(const uint4*)(ps + r * 32 + g * 8);
    ap.q[1] = *(const uint4*)(ps + r * 32 + 16 + g * 8);
    for (int c = 0; c < 4; ++c) {
      FragB bv;
      for (int j = 0; j < 16; ++j) bv.e[j] = vs[((j < 8) ? (g * 8 + j) : (16 + g * 8 + j - 8)) * 64 + c * 16 + r];
      o[c] = __builtin_amdgcn_wmma_f32_16x16x32_bf16(false, ap.v, false, bv.v, (short)0, o[c], false, false);
    }
  }

  for (int c = 0; c < 4; ++c)
    for (int rr = 0; rr < 8; ++rr) {
      int qrow = qt * 16 + rr + g * 8;
      float val = o[c][rr] / lrow[rr];
      Ob[(size_t)(b * S_ + qrow) * D_ + h * HD_ + c * 16 + r] = (bf16_t)val;
    }
}

// ---------------- host driver ----------------

extern "C" void kernel_launch(void* const* d_in, const int* in_sizes, int n_in,
                              void* d_out, int out_size, void* d_ws, size_t ws_size,
                              hipStream_t stream) {
  const float* tgt      = (const float*)d_in[0];
  const float* memoryp  = (const float*)d_in[1];
  const unsigned char* mask = (const unsigned char*)d_in[2];
  const float* w_in_sa  = (const float*)d_in[3];
  const float* b_in_sa  = (const float*)d_in[4];
  const float* w_out_sa = (const float*)d_in[5];
  const float* b_out_sa = (const float*)d_in[6];
  const float* w_in_ca  = (const float*)d_in[7];
  const float* b_in_ca  = (const float*)d_in[8];
  const float* w_out_ca = (const float*)d_in[9];
  const float* b_out_ca = (const float*)d_in[10];
  const float* w_ff1    = (const float*)d_in[11];
  const float* b_ff1    = (const float*)d_in[12];
  const float* w_ff2    = (const float*)d_in[13];
  const float* b_ff2    = (const float*)d_in[14];
  const float* ln1_g = (const float*)d_in[15];
  const float* ln1_b = (const float*)d_in[16];
  const float* ln2_g = (const float*)d_in[17];
  const float* ln2_b = (const float*)d_in[18];
  const float* ln3_g = (const float*)d_in[19];
  const float* ln3_b = (const float*)d_in[20];

  const int R = B_ * S_;                         // 8192 token rows
  char* ws = (char*)d_ws;
  auto alloc = [&](size_t bytes) -> void* {
    void* p = (void*)ws;
    ws += (bytes + 255) & ~(size_t)255;
    return p;
  };
  float*  x     = (float*)alloc((size_t)R * D_ * 4);
  bf16_t* hbuf  = (bf16_t*)alloc((size_t)R * D_ * 2);
  bf16_t* qkv   = (bf16_t*)alloc((size_t)R * 3 * D_ * 2);
  bf16_t* attnb = (bf16_t*)alloc((size_t)R * D_ * 2);
  bf16_t* qbuf  = (bf16_t*)alloc((size_t)R * D_ * 2);
  bf16_t* kbuf  = (bf16_t*)alloc((size_t)R * D_ * 2);
  bf16_t* vbuf  = (bf16_t*)alloc((size_t)R * D_ * 2);
  bf16_t* ffbuf = (bf16_t*)alloc((size_t)R * DFF_ * 2);
  bf16_t* memB  = (bf16_t*)alloc((size_t)B_ * M_ * D_ * 2);
  bf16_t* Wqkv_sa = (bf16_t*)alloc((size_t)L_ * 3 * D_ * D_ * 2);
  bf16_t* Wout_sa = (bf16_t*)alloc((size_t)L_ * D_ * D_ * 2);
  bf16_t* Wqkv_ca = (bf16_t*)alloc((size_t)L_ * 3 * D_ * D_ * 2);
  bf16_t* Wout_ca = (bf16_t*)alloc((size_t)L_ * D_ * D_ * 2);
  bf16_t* Wff1    = (bf16_t*)alloc((size_t)L_ * DFF_ * D_ * 2);
  bf16_t* Wff2    = (bf16_t*)alloc((size_t)L_ * D_ * DFF_ * 2);

  auto cvt = [&](const float* src, bf16_t* dst, size_t n) {
    k_f32_to_bf16<<<(unsigned)((n + 255) / 256), 256, 0, stream>>>(src, dst, (int)n);
  };
  cvt(w_in_sa,  Wqkv_sa, (size_t)L_ * 3 * D_ * D_);
  cvt(w_out_sa, Wout_sa, (size_t)L_ * D_ * D_);
  cvt(w_in_ca,  Wqkv_ca, (size_t)L_ * 3 * D_ * D_);
  cvt(w_out_ca, Wout_ca, (size_t)L_ * D_ * D_);
  cvt(w_ff1,    Wff1,    (size_t)L_ * DFF_ * D_);
  cvt(w_ff2,    Wff2,    (size_t)L_ * D_ * DFF_);

  k_tgt_to_x<<<(S_ * B_ * D_) / 256, 256, 0, stream>>>(tgt, x);
  k_mem_to_bsd<<<(M_ * B_ * D_) / 256, 256, 0, stream>>>(memoryp, memB);

  const int attnBlocks = (B_ * H_ * (S_ / 16)) / 8;   // 1024
  for (int i = 0; i < L_; ++i) {
    const size_t oDD = (size_t)i * D_ * D_;
    const size_t o3DD = (size_t)i * 3 * D_ * D_;
    // --- self attention (pre-norm) ---
    k_layernorm<<<R, 256, 0, stream>>>(x, ln1_g + i * D_, ln1_b + i * D_, hbuf);
    k_gemm<false, false><<<dim3(3 * D_ / 128, R / 128), 256, 0, stream>>>(
        hbuf, Wqkv_sa + o3DD, b_in_sa + (size_t)i * 3 * D_, nullptr, qkv, D_, 3 * D_);
    k_attn<true><<<attnBlocks, 256, 0, stream>>>(
        qkv, qkv + D_, qkv + 2 * D_, nullptr, attnb, 3 * D_, 3 * D_, S_);
    k_gemm<false, true><<<dim3(D_ / 128, R / 128), 256, 0, stream>>>(
        attnb, Wout_sa + oDD, b_out_sa + (size_t)i * D_, x, x, D_, D_);
    // --- cross attention ---
    k_layernorm<<<R, 256, 0, stream>>>(x, ln2_g + i * D_, ln2_b + i * D_, hbuf);
    k_gemm<false, false><<<dim3(D_ / 128, R / 128), 256, 0, stream>>>(
        hbuf, Wqkv_ca + o3DD, b_in_ca + (size_t)i * 3 * D_, nullptr, qbuf, D_, D_);
    k_gemm<false, false><<<dim3(D_ / 128, R / 128), 256, 0, stream>>>(
        memB, Wqkv_ca + o3DD + (size_t)D_ * D_, b_in_ca + (size_t)i * 3 * D_ + D_, nullptr, kbuf, D_, D_);
    k_gemm<false, false><<<dim3(D_ / 128, R / 128), 256, 0, stream>>>(
        memB, Wqkv_ca + o3DD + (size_t)2 * D_ * D_, b_in_ca + (size_t)i * 3 * D_ + 2 * D_, nullptr, vbuf, D_, D_);
    k_attn<false><<<attnBlocks, 256, 0, stream>>>(
        qbuf, kbuf, vbuf, mask, attnb, D_, D_, M_);
    k_gemm<false, true><<<dim3(D_ / 128, R / 128), 256, 0, stream>>>(
        attnb, Wout_ca + oDD, b_out_ca + (size_t)i * D_, x, x, D_, D_);
    // --- feed-forward ---
    k_layernorm<<<R, 256, 0, stream>>>(x, ln3_g + i * D_, ln3_b + i * D_, hbuf);
    k_gemm<true, false><<<dim3(DFF_ / 128, R / 128), 256, 0, stream>>>(
        hbuf, Wff1 + (size_t)i * DFF_ * D_, b_ff1 + (size_t)i * DFF_, nullptr, ffbuf, D_, DFF_);
    k_gemm<false, true><<<dim3(D_ / 128, R / 128), 256, 0, stream>>>(
        ffbuf, Wff2 + (size_t)i * D_ * DFF_, b_ff2 + (size_t)i * D_, x, x, DFF_, D_);
  }

  hipMemcpyAsync(d_out, x, (size_t)R * D_ * sizeof(float), hipMemcpyDeviceToDevice, stream);
}
